// PIDustModel_19344532702165
// MI455X (gfx1250) — matrix operand order repaired and checked by reading, not compiled, not run on hardware
//
#include <hip/hip_runtime.h>
#include <hip/hip_bf16.h>
#include <math.h>

typedef __attribute__((ext_vector_type(16))) _Float16 v16h;
typedef __attribute__((ext_vector_type(8)))  float    v8f;

#define PD_EPS 1e-8f

__global__ void pidust_init_delta(float* __restrict__ delta,
                                  const float* __restrict__ head_b, int N) {
  int i = blockIdx.x * blockDim.x + threadIdx.x;
  if (i < N) delta[i] = head_b[0];
}

// One wave processes tiles of 16 edges: builds a 16x32 f16 A-tile of edge
// features (K padded 10->32), multiplies against 4 register-resident 32x16
// B-tiles of mlp_w^T via v_wmma_f32_16x16x32_f16 with the bias pre-loaded
// into the C accumulator, applies ReLU, folds in head_w, reduces across the
// hidden dim with shfl_xor, and does one scalar atomicAdd per edge into
// delta[dst].
__global__ __launch_bounds__(256)
void pidust_edge_wmma(const float* __restrict__ x,        // (N,4)
                      const float* __restrict__ pos,      // (N,2)
                      const float* __restrict__ mlp_w,    // (64,10)
                      const float* __restrict__ mlp_b,    // (64)
                      const float* __restrict__ head_w,   // (64)
                      const long long* __restrict__ ei,   // (2,E) int64
                      float* __restrict__ delta,          // (N)
                      int E)
{
  const int lane = threadIdx.x & 31;
  const int half = lane >> 4;      // 0: K=0..15 of A/B, 1: K=16..31 (zero)
  const int l16  = lane & 15;

  const int waveInBlock   = threadIdx.x >> 5;
  const int wavesPerBlock = blockDim.x >> 5;
  const long long gwave   = (long long)blockIdx.x * wavesPerBlock + waveInBlock;
  const long long nwaves  = (long long)gridDim.x * wavesPerBlock;

  // ---- Build the 4 B-tiles (W^T, 32x16 each, K padded with zeros) ----
  // B layout: lanes 0-15 hold column N=l16, K=0..15 in the 16 halfs;
  // lanes 16-31 hold K=16..31 (all zero since fan_in=10).
  v16h B[4];
  v8f  Cbias[4];              // mlp_b broadcast into the C/D column layout
  float hw[4];
#pragma unroll
  for (int g = 0; g < 4; ++g) {
    const int col = 16 * g + l16;           // hidden unit index 0..63
    v16h b = {};
    if (half == 0) {
#pragma unroll
      for (int k = 0; k < 10; ++k)
        b[k] = (_Float16)mlp_w[col * 10 + k];
    }
    B[g] = b;
    hw[g] = head_w[col];
    const float mb = mlp_b[col];
    v8f cb;
#pragma unroll
    for (int r = 0; r < 8; ++r) cb[r] = mb;
    Cbias[g] = cb;
  }

  const long long numTiles = ((long long)E + 15) >> 4;

  for (long long t = gwave; t < numTiles; t += nwaves) {
    const long long base = t << 4;
    const long long e    = base + l16;
    const bool valid     = (e < (long long)E);
    const long long ec   = valid ? e : (long long)(E - 1);

    const int s = (int)ei[ec];                    // src node
    const int d = (int)ei[(long long)E + ec];     // dst node

    const float4 xs = ((const float4*)x)[s];
    const float4 xd = ((const float4*)x)[d];
    const float2 ps = ((const float2*)pos)[s];
    const float2 pd = ((const float2*)pos)[d];

    const float dx   = ps.x - pd.x;
    const float dy   = ps.y - pd.y;
    const float dist = sqrtf(dx * dx + dy * dy + PD_EPS);
    const float inv  = 1.0f / dist;
    const float wind = (xs.y * dx + xs.z * dy) * inv;

    // ---- Pack A fragment (16x32 f16, rows = edges in tile) ----
    v16h A = {};
    if (valid) {
      if (half == 0) {
        // K = 0..9 : [x_dst(4), x_src(4), wind_proj, dist]; K=10..15 zero pad
        A[0] = (_Float16)xd.x; A[1] = (_Float16)xd.y;
        A[2] = (_Float16)xd.z; A[3] = (_Float16)xd.w;
        A[4] = (_Float16)xs.x; A[5] = (_Float16)xs.y;
        A[6] = (_Float16)xs.z; A[7] = (_Float16)xs.w;
        A[8] = (_Float16)wind; A[9] = (_Float16)dist;
      }
      // lanes 16-31 (K=16..31) stay zero
    }

    // ---- 4 WMMAs covering hidden columns 0..63; bias rides in C ----
    // D layout: lane holds column 16g + l16; VGPR r -> row M = 8*half + r.
    float acc[8] = {0.f, 0.f, 0.f, 0.f, 0.f, 0.f, 0.f, 0.f};
#pragma unroll
    for (int g = 0; g < 4; ++g) {
      v8f c = __builtin_amdgcn_wmma_f32_16x16x32_f16(
                  /*neg_a=*/false, A, /*neg_b=*/false, B[g],
                  /*c_mod=*/(short)0, Cbias[g],
                  /*reuse_a=*/false, /*reuse_b=*/false);
#pragma unroll
      for (int r = 0; r < 8; ++r) {
        float m = c[r] > 0.f ? c[r] : 0.f;     // ReLU (bias already in c)
        acc[r] = fmaf(m, hw[g], acc[r]);       // fold head_w
      }
    }

    // ---- Sum over hidden dim: butterfly across the 16 lanes of each half ----
#pragma unroll
    for (int r = 0; r < 8; ++r) {
#pragma unroll
      for (int m = 1; m < 16; m <<= 1)
        acc[r] += __shfl_xor(acc[r], m, 32);   // masks <16 stay within half
    }

    // dst for edge (8*half + l16) lives in lane (8*half + l16) — no reload.
    const int dn_shuf = __shfl(d, 8 * half + l16, 32);

    // ---- One atomic per edge: lanes l16<8 own edge m = 8*half + l16 ----
    if (l16 < 8) {
      const int j = l16;
      float v = acc[0];
#pragma unroll
      for (int r = 1; r < 8; ++r) v = (j == r) ? acc[r] : v;
      const long long eo = base + (long long)(8 * half + j);
      if (eo < (long long)E) {
        atomicAdd(&delta[dn_shuf], v);
      }
    }
  }
}

__global__ void pidust_finalize(const float* __restrict__ x,
                                const float* __restrict__ delta,
                                float* __restrict__ pm10, int N) {
  int i = blockIdx.x * blockDim.x + threadIdx.x;
  if (i < N) {
    const float t = x[i * 4 + 0] + delta[i];
    // numerically stable softplus
    const float sp = (t > 0.f) ? (t + log1pf(expf(-t))) : log1pf(expf(t));
    pm10[i] = sp;
  }
}

extern "C" void kernel_launch(void* const* d_in, const int* in_sizes, int n_in,
                              void* d_out, int out_size, void* d_ws, size_t ws_size,
                              hipStream_t stream) {
  (void)n_in; (void)out_size; (void)d_ws; (void)ws_size;

  const float*     x      = (const float*)d_in[0];
  const float*     pos    = (const float*)d_in[1];
  const float*     mlp_w  = (const float*)d_in[2];
  const float*     mlp_b  = (const float*)d_in[3];
  const float*     head_w = (const float*)d_in[4];
  const float*     head_b = (const float*)d_in[5];
  const long long* ei     = (const long long*)d_in[6];

  const int N = in_sizes[0] / 4;
  const int E = in_sizes[6] / 2;

  float* out   = (float*)d_out;
  float* pm10  = out;       // first N floats
  float* delta = out + N;   // second N floats (accumulated in place)

  // 1) delta = head_b (fresh every launch; harness does not re-poison)
  pidust_init_delta<<<(N + 255) / 256, 256, 0, stream>>>(delta, head_b, N);

  // 2) fused edge MLP (WMMA) + scalar scatter-add
  const long long numTiles = ((long long)E + 15) / 16;
  const int wavesPerBlock  = 8;  // 256 threads = 8 wave32
  long long wantBlocks = (numTiles + wavesPerBlock - 1) / wavesPerBlock;
  int blocks = (int)(wantBlocks < 4096 ? wantBlocks : 4096);
  if (blocks < 1) blocks = 1;
  pidust_edge_wmma<<<blocks, 256, 0, stream>>>(x, pos, mlp_w, mlp_b, head_w,
                                               ei, delta, E);

  // 3) pm10 = softplus(x[:,0] + delta)
  pidust_finalize<<<(N + 255) / 256, 256, 0, stream>>>(x, delta, pm10, N);
}
